// DeformNet2_833223656366
// MI455X (gfx1250) — matrix operand order repaired and checked by reading, not compiled, not run on hardware
//
#include <hip/hip_runtime.h>
#include <hip/hip_bf16.h>

typedef __attribute__((ext_vector_type(16))) _Float16 v16h;
typedef __attribute__((ext_vector_type(8)))  float    v8f;

#define SPAT 6272          // 28*28*8
#define NPOS 25088         // 4*SPAT
#define BNEPS 1e-5f
#define AROW 36            // padded LDS A row stride (halves)

// ---- workspace layout (bytes) ----
#define OFF_Z1    0UL
#define OFF_Z2    3211264UL
#define OFF_Z3    9633792UL
#define OFF_Z4    22478848UL
#define OFF_BP2   35323904UL
#define OFF_BP3   35434496UL
#define OFF_BPO   35876864UL
#define OFF_BPD   36540416UL
#define OFF_STATS 37425152UL   // 1024 floats: sum[4][128], sq[4][128]
#define OFF_BN    37429248UL   // 1024 floats: a[4][128], b[4][128]
#define OFF_POOL  37433344UL   // 512 floats

__global__ void __launch_bounds__(256) zero_kernel(float* p, int n) {
    int i = blockIdx.x * 256 + threadIdx.x;
    if (i < n) p[i] = 0.f;
}

// Pack f32 weights (O, Cin, 27) into fragment-major f16 B tiles.
// K order: K = t*Cin + c  (Cin = power of two -> shift/mask decode)
__global__ void __launch_bounds__(256) pack_kernel(const float* __restrict__ w,
                                                   _Float16* __restrict__ outp,
                                                   int O, int Cin, int cshift, int NT) {
    int idx = blockIdx.x * 256 + threadIdx.x;
    int KT = (Cin * 27) >> 5;
    int total = KT * NT * 512;
    if (idx >= total) return;
    int j    = idx & 15;
    int lane = (idx >> 4) & 31;
    int ntk  = idx >> 9;
    int nt   = ntk % NT;
    int kt   = ntk / NT;
    int K = kt * 32 + ((lane >> 4) ? 16 : 0) + j;
    int N = nt * 16 + (lane & 15);
    float v = 0.f;
    if (N < O) {
        int c = K & (Cin - 1), t = K >> cshift;
        v = w[((size_t)N * Cin + c) * 27 + t];
    }
    outp[idx] = (_Float16)v;
}

// conv1: Cin=1, direct; relu; channel-last z1 (pos, 32); per-channel stats.
__global__ void __launch_bounds__(256) conv1_kernel(const float* __restrict__ x,
                                                    const float* __restrict__ w,
                                                    const float* __restrict__ bias,
                                                    float* __restrict__ z1,
                                                    float* __restrict__ sSum,
                                                    float* __restrict__ sSq) {
    __shared__ float wsh[864];        // [t][o]
    __shared__ float lsum[32], lsq[32];
    int tid = threadIdx.x;
    int o = tid & 31;
    for (int i = tid; i < 864; i += 256) { int t = i >> 5, oo = i & 31; wsh[i] = w[oo * 27 + t]; }
    if (tid < 32) { lsum[tid] = 0.f; lsq[tid] = 0.f; }
    __syncthreads();
    int p  = blockIdx.x * 8 + (tid >> 5);
    int w_ = p & 7, h_ = (p >> 3) % 28, d_ = (p / 224) % 28, b_ = p / 6272;
    const float* xb = x + (size_t)b_ * SPAT;
    float acc = bias[o];
#pragma unroll
    for (int t = 0; t < 27; t++) {
        int zd = d_ + t / 9 - 1, zh = h_ + (t / 3) % 3 - 1, zw = w_ + t % 3 - 1;
        if ((unsigned)zd < 28u && (unsigned)zh < 28u && (unsigned)zw < 8u)
            acc = fmaf(xb[zd * 224 + zh * 8 + zw], wsh[t * 32 + o], acc);
    }
    acc = acc > 0.f ? acc : 0.f;
    z1[(size_t)p * 32 + o] = acc;     // channel-last
    atomicAdd(&lsum[o], acc);
    atomicAdd(&lsq[o], acc * acc);
    __syncthreads();
    if (tid < 32) { atomicAdd(&sSum[tid], lsum[tid]); atomicAdd(&sSq[tid], lsq[tid]); }
}

__global__ void __launch_bounds__(128) bn_finalize_kernel(const float* __restrict__ sSum,
                                                          const float* __restrict__ sSq,
                                                          const float* __restrict__ g,
                                                          const float* __restrict__ beta,
                                                          float* __restrict__ a,
                                                          float* __restrict__ bb, int C) {
    int c = threadIdx.x;
    if (c >= C) return;
    float m  = sSum[c] / (float)NPOS;
    float v  = sSq[c] / (float)NPOS - m * m;
    float ia = g[c] * rsqrtf(v + BNEPS);
    a[c] = ia;
    bb[c] = beta[c] - m * ia;
}

// Generic im2col + WMMA conv; channel-last input, BN folded on load.
// NW waves/block, tile = 16 positions x NW*16 out-channels; double-buffered A in LDS,
// K loop unrolled by 3 (KT % 3 == 0 for all layers). Branchless boundary handling.
// MODE 0: out = relu(gemm+bias), channel-last (pos, O) + BN stats
// MODE 1: offsets, store gemm+bias (o<O) to (B,81,D,H,W) channel-first
template <int CIN, int CSHIFT, int O, int NW, int MODE>
__global__ void __launch_bounds__(NW * 32)
conv_gemm_kernel(const float* __restrict__ xin, const float* __restrict__ bnA,
                 const float* __restrict__ bnB, const _Float16* __restrict__ bpack,
                 const float* __restrict__ bias, float* __restrict__ outp,
                 float* __restrict__ sSum, float* __restrict__ sSq) {
    constexpr int KT  = (CIN * 27) >> 5;
    constexpr int NTH = NW * 32;
    __shared__ _Float16 As[2][16 * AROW];
    __shared__ int   sOff[432];       // [t*16+m] neighbor spatial offset or -1
    __shared__ int   bIdx[16];
    __shared__ float bnAs[CIN], bnBs[CIN];
    __shared__ float lsum[128], lsq[128];
    int tid = threadIdx.x, wave = tid >> 5, lane = tid & 31;
    int ptile = blockIdx.x;
    if (MODE == 0 && tid < 128) { lsum[tid] = 0.f; lsq[tid] = 0.f; }
    for (int i = tid; i < CIN; i += NTH) { bnAs[i] = bnA[i]; bnBs[i] = bnB[i]; }
    if (tid < 16) bIdx[tid] = (ptile * 16 + tid) / 6272;
    for (int e = tid; e < 432; e += NTH) {
        int t = e >> 4, m = e & 15;
        int p  = ptile * 16 + m;
        int w_ = p & 7, h_ = (p >> 3) % 28, d_ = (p / 224) % 28;
        int zd = d_ + t / 9 - 1, zh = h_ + (t / 3) % 3 - 1, zw = w_ + t % 3 - 1;
        sOff[e] = ((unsigned)zd < 28u && (unsigned)zh < 28u && (unsigned)zw < 8u)
                      ? (zd * 224 + zh * 8 + zw) : -1;
    }
    int m4  = tid >> 3;               // builder: 128 threads, one float4 each
    int kk4 = (tid & 7) << 2;
    auto buildA = [&](int kt, _Float16* dst) {
        if (tid < 128) {
            int kg = kt * 32 + kk4;
            int c  = kg & (CIN - 1);
            int t  = kg >> CSHIFT;
            int so = sOff[t * 16 + m4];
            bool ok = so >= 0;
            int soc = ok ? so : 0;
            const float4 z = *reinterpret_cast<const float4*>(
                xin + ((size_t)(bIdx[m4] * SPAT + soc)) * CIN + c);
            _Float16* d = dst + m4 * AROW + kk4;
            d[0] = (_Float16)(ok ? fmaf(z.x, bnAs[c],     bnBs[c])     : 0.f);
            d[1] = (_Float16)(ok ? fmaf(z.y, bnAs[c + 1], bnBs[c + 1]) : 0.f);
            d[2] = (_Float16)(ok ? fmaf(z.z, bnAs[c + 2], bnBs[c + 2]) : 0.f);
            d[3] = (_Float16)(ok ? fmaf(z.w, bnAs[c + 3], bnBs[c + 3]) : 0.f);
        }
    };
    __syncthreads();                  // LUTs ready
    buildA(0, As[0]);
    v8f acc = {0.f, 0.f, 0.f, 0.f, 0.f, 0.f, 0.f, 0.f};
    int kb = (lane >> 4) << 4;
    for (int kt0 = 0; kt0 < KT; kt0 += 3) {
#pragma unroll
        for (int u = 0; u < 3; u++) {
            const int kt = kt0 + u;
            __syncthreads();          // buf[kt&1] built; prior reads done
            const _Float16* row = &As[kt & 1][(lane & 15) * AROW];
            v16h af;
#pragma unroll
            for (int j = 0; j < 8; j++) { af[j] = row[kb + j]; af[8 + j] = row[kb + 8 + j]; }
            v16h bf = *reinterpret_cast<const v16h*>(
                bpack + (((size_t)(kt * NW + wave) * 32 + lane) << 4));
            if (kt + 1 < KT) buildA(kt + 1, As[(kt + 1) & 1]);
            acc = __builtin_amdgcn_wmma_f32_16x16x32_f16(false, af, false, bf, (short)0,
                                                         acc, false, false);
        }
    }
    int ncol  = wave * 16 + (lane & 15);
    int mbase = (lane >> 4) << 3;
    float bo = (ncol < O) ? bias[ncol] : 0.f;
    float ms = 0.f, mq = 0.f;
#pragma unroll
    for (int r = 0; r < 8; r++) {
        int p = ptile * 16 + mbase + r;
        float v = acc[r] + bo;
        if (MODE == 0) {
            v = v > 0.f ? v : 0.f;
            ms += v; mq += v * v;
            outp[(size_t)p * O + ncol] = v;   // channel-last, coalesced
        } else if (ncol < O) {
            int w_ = p & 7, h_ = (p >> 3) % 28, d_ = (p / 224) % 28, b_ = p / 6272;
            outp[((size_t)(b_ * 81 + ncol)) * SPAT + d_ * 224 + h_ * 8 + w_] = v;
        }
    }
    if (MODE == 0) {
        atomicAdd(&lsum[wave * 16 + (lane & 15)], ms);
        atomicAdd(&lsq[wave * 16 + (lane & 15)], mq);
        __syncthreads();
        if (tid < O) { atomicAdd(&sSum[tid], lsum[tid]); atomicAdd(&sSq[tid], lsq[tid]); }
    }
}

// Deformable conv: trilinear-gathered im2col + WMMA. Cin=O=128, 8 waves, grid 1568.
// Channel-last xin/out; per-(pos,tap) 8 corner byte-offsets + weights precomputed in LDS
// in AoS layout (two b128 LDS loads each); K-loop gather = 8 branchless float2 loads.
// BN factored out of gather: a*Sum(w z) + b*Sum_valid(w).
__global__ void __launch_bounds__(256)
deform_gemm_kernel(const float* __restrict__ xin, const float* __restrict__ bnA,
                   const float* __restrict__ bnB, const float* __restrict__ offs,
                   const _Float16* __restrict__ bpack, const float* __restrict__ bias,
                   float* __restrict__ outp, float* __restrict__ sSum,
                   float* __restrict__ sSq) {
    constexpr int CIN = 128, O = 128, NW = 8, KT = 108;
    __shared__ _Float16 As[2][16 * AROW];
    __shared__ int   cOff[432][8];    // pre-clamped BYTE offsets (already *CIN*4)
    __shared__ float cWgt[432][8];    // corner weights (0 for invalid)
    __shared__ float wSum[432];       // sum of valid weights (for BN beta term)
    __shared__ int   bIdx[16];
    __shared__ float bnAs[128], bnBs[128];
    __shared__ float lsum[128], lsq[128];
    int tid = threadIdx.x, wave = tid >> 5, lane = tid & 31;
    int ptile = blockIdx.x;
    if (tid < 128) {
        lsum[tid] = 0.f; lsq[tid] = 0.f;
        bnAs[tid] = bnA[tid]; bnBs[tid] = bnB[tid];
    }
    if (tid < 16) bIdx[tid] = (ptile * 16 + tid) / 6272;
    // per-(tap, position): expand trilinear sampling into 8 byte-offsets + weights
    for (int e = tid; e < 432; e += 256) {
        int n = e >> 4, m = e & 15;
        int p = ptile * 16 + m;
        int w_ = p & 7, h_ = (p >> 3) % 28, d_ = (p / 224) % 28, b_ = p / 6272;
        size_t sb = (size_t)b_ * 81 * SPAT + d_ * 224 + h_ * 8 + w_;
        float pd = (float)(d_ + n / 9)       + offs[sb + (size_t)n * SPAT];
        float ph = (float)(h_ + (n / 3) % 3) + offs[sb + (size_t)(27 + n) * SPAT];
        float pw = (float)(w_ + n % 3)       + offs[sb + (size_t)(54 + n) * SPAT];
        pd = fminf(fmaxf(pd, 0.f), 29.f);
        ph = fminf(fmaxf(ph, 0.f), 29.f);
        pw = fminf(fmaxf(pw, 0.f), 9.f);
        float qdf = fminf(floorf(pd), 28.f);
        float qhf = fminf(floorf(ph), 28.f);
        float qwf = fminf(floorf(pw), 8.f);
        float td = fminf(fmaxf(pd - qdf, 0.f), 1.f);
        float th = fminf(fmaxf(ph - qhf, 0.f), 1.f);
        float tw = fminf(fmaxf(pw - qwf, 0.f), 1.f);
        int qd = (int)qdf, qh = (int)qhf, qw = (int)qwf;
        float ws = 0.f;
#pragma unroll
        for (int corner = 0; corner < 8; corner++) {
            int i = corner >> 2, jj = (corner >> 1) & 1, k2 = corner & 1;
            int od = qd + i - 1, oh = qh + jj - 1, ow = qw + k2 - 1;
            bool ok = (unsigned)od < 28u && (unsigned)oh < 28u && (unsigned)ow < 8u;
            float wgt = (i ? td : 1.f - td) * (jj ? th : 1.f - th) * (k2 ? tw : 1.f - tw);
            wgt = ok ? wgt : 0.f;
            cOff[e][corner] = ok ? (od * 224 + oh * 8 + ow) * (CIN * 4) : 0;  // bytes
            cWgt[e][corner] = wgt;
            ws += wgt;
        }
        wSum[e] = ws;
    }
    int m2  = tid >> 4;               // builder: 256 threads, one float2 each
    int kk2 = (tid & 15) << 1;
    auto buildA = [&](int kt, _Float16* dst) {
        int kg = kt * 32 + kk2;
        int c0 = kg & (CIN - 1);
        int n  = kg >> 7;
        int si = n * 16 + m2;
        const char* base = (const char*)(xin + ((size_t)bIdx[m2] * SPAT) * CIN + c0);
        const int4*   op = reinterpret_cast<const int4*>(&cOff[si][0]);
        const float4* wp = reinterpret_cast<const float4*>(&cWgt[si][0]);
        int4   o0 = op[0], o1 = op[1];
        float4 w0 = wp[0], w1 = wp[1];
        float gx = 0.f, gy = 0.f;
        float2 z;
        z = *reinterpret_cast<const float2*>(base + o0.x);
        gx = fmaf(w0.x, z.x, gx); gy = fmaf(w0.x, z.y, gy);
        z = *reinterpret_cast<const float2*>(base + o0.y);
        gx = fmaf(w0.y, z.x, gx); gy = fmaf(w0.y, z.y, gy);
        z = *reinterpret_cast<const float2*>(base + o0.z);
        gx = fmaf(w0.z, z.x, gx); gy = fmaf(w0.z, z.y, gy);
        z = *reinterpret_cast<const float2*>(base + o0.w);
        gx = fmaf(w0.w, z.x, gx); gy = fmaf(w0.w, z.y, gy);
        z = *reinterpret_cast<const float2*>(base + o1.x);
        gx = fmaf(w1.x, z.x, gx); gy = fmaf(w1.x, z.y, gy);
        z = *reinterpret_cast<const float2*>(base + o1.y);
        gx = fmaf(w1.y, z.x, gx); gy = fmaf(w1.y, z.y, gy);
        z = *reinterpret_cast<const float2*>(base + o1.z);
        gx = fmaf(w1.z, z.x, gx); gy = fmaf(w1.z, z.y, gy);
        z = *reinterpret_cast<const float2*>(base + o1.w);
        gx = fmaf(w1.w, z.x, gx); gy = fmaf(w1.w, z.y, gy);
        float wv = wSum[si];
        _Float16* d = dst + m2 * AROW + kk2;
        d[0] = (_Float16)fmaf(gx, bnAs[c0],     bnBs[c0]     * wv);
        d[1] = (_Float16)fmaf(gy, bnAs[c0 + 1], bnBs[c0 + 1] * wv);
    };
    __syncthreads();                  // sampling LUTs ready
    buildA(0, As[0]);
    v8f acc = {0.f, 0.f, 0.f, 0.f, 0.f, 0.f, 0.f, 0.f};
    int kb = (lane >> 4) << 4;
    for (int kt0 = 0; kt0 < KT; kt0 += 3) {
#pragma unroll
        for (int u = 0; u < 3; u++) {
            const int kt = kt0 + u;
            __syncthreads();
            const _Float16* row = &As[kt & 1][(lane & 15) * AROW];
            v16h af;
#pragma unroll
            for (int j = 0; j < 8; j++) { af[j] = row[kb + j]; af[8 + j] = row[kb + 8 + j]; }
            v16h bf = *reinterpret_cast<const v16h*>(
                bpack + (((size_t)(kt * NW + wave) * 32 + lane) << 4));
            if (kt + 1 < KT) buildA(kt + 1, As[(kt + 1) & 1]);
            acc = __builtin_amdgcn_wmma_f32_16x16x32_f16(false, af, false, bf, (short)0,
                                                         acc, false, false);
        }
    }
    int ncol  = wave * 16 + (lane & 15);
    int mbase = (lane >> 4) << 3;
    float bo = bias[ncol];
    float ms = 0.f, mq = 0.f;
#pragma unroll
    for (int r = 0; r < 8; r++) {
        int p = ptile * 16 + mbase + r;
        float v = acc[r] + bo;
        v = v > 0.f ? v : 0.f;
        ms += v; mq += v * v;
        outp[(size_t)p * O + ncol] = v;   // channel-last
    }
    atomicAdd(&lsum[wave * 16 + (lane & 15)], ms);
    atomicAdd(&lsq[wave * 16 + (lane & 15)], mq);
    __syncthreads();
    if (tid < 128) { atomicAdd(&sSum[tid], lsum[tid]); atomicAdd(&sSq[tid], lsq[tid]); }
}

// channel-last pool: grid=4 (batch), block=128 (channel); fully coalesced
__global__ void __launch_bounds__(128) pool_kernel(const float* __restrict__ z4,
                                                   float* __restrict__ pooled) {
    int b = blockIdx.x, c = threadIdx.x;
    const float* p = z4 + (size_t)b * SPAT * 128 + c;
    float s = 0.f;
    for (int i = 0; i < SPAT; i++) s += p[(size_t)i * 128];
    pooled[b * 128 + c] = s / (float)SPAT;
}

__global__ void __launch_bounds__(128) head_kernel(const float* __restrict__ pooled,
                                                   const float* __restrict__ a4,
                                                   const float* __restrict__ bb4,
                                                   const float* __restrict__ fcw,
                                                   const float* __restrict__ fcb,
                                                   float* __restrict__ outp) {
    __shared__ float pl[4][128];
    __shared__ float lg[4][10];
    int tid = threadIdx.x;
    for (int b = 0; b < 4; b++) pl[b][tid] = fmaf(pooled[b * 128 + tid], a4[tid], bb4[tid]);
    __syncthreads();
    if (tid < 40) {
        int b = tid / 10, o = tid % 10;
        float s = fcb[o];
        for (int c = 0; c < 128; c++) s = fmaf(pl[b][c], fcw[o * 128 + c], s);
        lg[b][o] = s;
    }
    __syncthreads();
    if (tid < 4) {
        int b = tid;
        float mx = lg[b][0];
        for (int o = 1; o < 10; o++) mx = fmaxf(mx, lg[b][o]);
        float se = 0.f;
        for (int o = 0; o < 10; o++) se += expf(lg[b][o] - mx);
        float ls = logf(se);
        for (int o = 0; o < 10; o++) outp[b * 10 + o] = lg[b][o] - mx - ls;
    }
}

extern "C" void kernel_launch(void* const* d_in, const int* in_sizes, int n_in,
                              void* d_out, int out_size, void* d_ws, size_t ws_size,
                              hipStream_t stream) {
    const float* x   = (const float*)d_in[0];
    const float* c1w = (const float*)d_in[1];
    const float* c1b = (const float*)d_in[2];
    const float* g1  = (const float*)d_in[3];
    const float* b1  = (const float*)d_in[4];
    const float* c2w = (const float*)d_in[5];
    const float* c2b = (const float*)d_in[6];
    const float* g2  = (const float*)d_in[7];
    const float* b2  = (const float*)d_in[8];
    const float* c3w = (const float*)d_in[9];
    const float* c3b = (const float*)d_in[10];
    const float* g3  = (const float*)d_in[11];
    const float* b3  = (const float*)d_in[12];
    const float* ow_ = (const float*)d_in[13];
    const float* ob  = (const float*)d_in[14];
    const float* dw  = (const float*)d_in[15];
    const float* db  = (const float*)d_in[16];
    const float* g4  = (const float*)d_in[17];
    const float* b4  = (const float*)d_in[18];
    const float* fcw = (const float*)d_in[19];
    const float* fcb = (const float*)d_in[20];

    float* outp = (float*)d_out;
    float* offsets_out = outp + 40;

    char* ws = (char*)d_ws;
    float*    z1  = (float*)(ws + OFF_Z1);
    float*    z2  = (float*)(ws + OFF_Z2);
    float*    z3  = (float*)(ws + OFF_Z3);
    float*    z4  = (float*)(ws + OFF_Z4);
    _Float16* bp2 = (_Float16*)(ws + OFF_BP2);
    _Float16* bp3 = (_Float16*)(ws + OFF_BP3);
    _Float16* bpo = (_Float16*)(ws + OFF_BPO);
    _Float16* bpd = (_Float16*)(ws + OFF_BPD);
    float* stats  = (float*)(ws + OFF_STATS);
    float* bn     = (float*)(ws + OFF_BN);
    float* pooled = (float*)(ws + OFF_POOL);

    float* sum0 = stats + 0 * 128; float* sq0 = stats + 512 + 0 * 128;
    float* sum1 = stats + 1 * 128; float* sq1 = stats + 512 + 1 * 128;
    float* sum2 = stats + 2 * 128; float* sq2 = stats + 512 + 2 * 128;
    float* sum3 = stats + 3 * 128; float* sq3 = stats + 512 + 3 * 128;
    float* a0 = bn + 0 * 128; float* bb0 = bn + 512 + 0 * 128;
    float* a1 = bn + 1 * 128; float* bb1 = bn + 512 + 1 * 128;
    float* a2 = bn + 2 * 128; float* bb2 = bn + 512 + 2 * 128;
    float* a3 = bn + 3 * 128; float* bb3 = bn + 512 + 3 * 128;

    zero_kernel<<<4, 256, 0, stream>>>(stats, 1024);

    pack_kernel<<<216, 256, 0, stream>>>(c2w, bp2, 64, 32, 5, 4);
    pack_kernel<<<864, 256, 0, stream>>>(c3w, bp3, 128, 64, 6, 8);
    pack_kernel<<<1296, 256, 0, stream>>>(ow_, bpo, 81, 128, 7, 6);
    pack_kernel<<<1728, 256, 0, stream>>>(dw, bpd, 128, 128, 7, 8);

    conv1_kernel<<<3136, 256, 0, stream>>>(x, c1w, c1b, z1, sum0, sq0);
    bn_finalize_kernel<<<1, 128, 0, stream>>>(sum0, sq0, g1, b1, a0, bb0, 32);

    conv_gemm_kernel<32, 5, 64, 4, 0><<<1568, 128, 0, stream>>>(z1, a0, bb0, bp2, c2b,
                                                                z2, sum1, sq1);
    bn_finalize_kernel<<<1, 128, 0, stream>>>(sum1, sq1, g2, b2, a1, bb1, 64);

    conv_gemm_kernel<64, 6, 128, 8, 0><<<1568, 256, 0, stream>>>(z2, a1, bb1, bp3, c3b,
                                                                 z3, sum2, sq2);
    bn_finalize_kernel<<<1, 128, 0, stream>>>(sum2, sq2, g3, b3, a2, bb2, 128);

    // offsets conv (output #2 of the reference, written straight to d_out+40)
    conv_gemm_kernel<128, 7, 81, 6, 1><<<1568, 192, 0, stream>>>(z3, a2, bb2, bpo, ob,
                                                                 offsets_out, nullptr,
                                                                 nullptr);

    deform_gemm_kernel<<<1568, 256, 0, stream>>>(z3, a2, bb2, offsets_out, bpd, db,
                                                 z4, sum3, sq3);
    bn_finalize_kernel<<<1, 128, 0, stream>>>(sum3, sq3, g4, b4, a3, bb3, 128);

    pool_kernel<<<4, 128, 0, stream>>>(z4, pooled);
    head_kernel<<<1, 128, 0, stream>>>(pooled, a3, bb3, fcw, fcb, outp);
}
